// CausalUserEncoder_68925635166852
// MI455X (gfx1250) — compile-verified, matrix operand order checked by reading, not run on hardware
//
#include <hip/hip_runtime.h>

typedef __attribute__((ext_vector_type(16))) _Float16 v16h;
typedef __attribute__((ext_vector_type(8)))  _Float16 v8h;
typedef __attribute__((ext_vector_type(8)))  float    v8f;
typedef __attribute__((ext_vector_type(4)))  float    v4f;
typedef __attribute__((ext_vector_type(4)))  int      v4i;

// async-to-LDS builtin takes v4i pointers in AS1 (global) / AS3 (LDS)
using as1_v4i = __attribute__((address_space(1))) v4i;
using as3_v4i = __attribute__((address_space(3))) v4i;

#define Dm   256
#define Ltok 2048
#define Bb   64
#define Hh   128
#define DUu  1024
#define NTOK (Bb * Ltok)
#define LDSROW 528   /* 512B row + 16B pad -> bank-conflict-free b128 reads */

#if defined(__has_builtin)
#if __has_builtin(__builtin_amdgcn_global_load_async_to_lds_b128) && \
    __has_builtin(__builtin_amdgcn_s_wait_asynccnt)
#define HAS_ASYNC 1
#endif
#endif
#ifndef HAS_ASYNC
#define HAS_ASYNC 0
#endif

__device__ __forceinline__ v8f wmma_f16(v16h a, v16h b, v8f c) {
  // D = A(16x32 f16) * B(32x16 f16) + C(16x16 f32)
  return __builtin_amdgcn_wmma_f32_16x16x32_f16(false, a, false, b, (short)0, c,
                                                false, false);
}

// ---------------------------------------------------------------------------
// K0a: convert Wa (1024x256) and Ws1 first half (128x256 of the 128x512) to f16
// ---------------------------------------------------------------------------
__global__ void convert_weights(const float* __restrict__ Wa,
                                const float* __restrict__ Ws1,
                                _Float16* __restrict__ wa16,
                                _Float16* __restrict__ ws1a16) {
  int i = blockIdx.x * blockDim.x + threadIdx.x;
  const int nWa = DUu * Dm;
  if (i < nWa) {
    wa16[i] = (_Float16)Wa[i];
  } else {
    int j = i - nWa;
    if (j < Hh * Dm) {
      int row = j >> 8;
      int col = j & 255;
      ws1a16[j] = (_Float16)Ws1[row * (2 * Dm) + col];  // first 256 cols only
    }
  }
}

// ---------------------------------------------------------------------------
// K0b: hvb[h] = bs1[h] + sum_d Ws1[h][256+d] * virtual_node[d]
//      (the virtual-node half of the struct-head GEMM is token-independent)
// ---------------------------------------------------------------------------
__global__ void virt_bias(const float* __restrict__ Ws1,
                          const float* __restrict__ bs1,
                          const float* __restrict__ vnode,
                          float* __restrict__ hvb) {
  int h = threadIdx.x;  // 128 threads
  float s = bs1[h];
  for (int d = 0; d < Dm; ++d) s += Ws1[h * (2 * Dm) + Dm + d] * vnode[d];
  hvb[h] = s;
}

// ---------------------------------------------------------------------------
// K1: per wave = 16 tokens.
//  - build h_events rows in f16 A-fragment layout (and mirror to global ws)
//  - struct head: 8 n-tiles x 8 K-step WMMAs vs Ws1-f16, leaky-relu, Ws2 dot,
//    cross-lane reduce, sigmoid(logit/0.01)*mask -> A_hat (written to d_out)
// ---------------------------------------------------------------------------
__global__ __launch_bounds__(256) void prep_struct_kernel(
    const int* __restrict__ q_seq, const int* __restrict__ r_seq,
    const float* __restrict__ t_seq, const float* __restrict__ mask,
    const float* __restrict__ q_tab, const float* __restrict__ r_tab,
    const _Float16* __restrict__ ws1a16, const float* __restrict__ hvb,
    const float* __restrict__ Ws2, const float* __restrict__ bs2,
    _Float16* __restrict__ h16, float* __restrict__ ahat) {
  const int wave = threadIdx.x >> 5;
  const int lane = threadIdx.x & 31;
  const int r    = lane & 15;   // matrix row within tile (M)
  const int kh   = lane >> 4;   // K-half selector per ISA A layout
  const int tok0 = (blockIdx.x * 8 + wave) * 16;
  const int tok  = tok0 + r;

  const int   q  = q_seq[tok];
  const int   rr = r_seq[tok];
  const float t  = t_seq[tok];

  const float* qrow = q_tab + (long)q * Dm;
  const float* rrow = r_tab + (long)rr * Dm;

  const float NEG_LN = -0.07195578414981393f;  // -ln(10000)/128

  // A-fragment layout (16-bit A 16x32): lane r = M; VGPRs 0-3 = K kh*8+{0..7},
  // VGPRs 4-7 = K 16+kh*8+{0..7}.  Each 8-element run is contiguous in K, so
  // it doubles as a coalescible v8h store of the f16 h_events workspace.
  v16h afrag[8];
#pragma unroll
  for (int s = 0; s < 8; ++s) {
#pragma unroll
    for (int h2 = 0; h2 < 2; ++h2) {
      const int c0 = s * 32 + h2 * 16 + kh * 8;
      v4f qa = *(const v4f*)(qrow + c0);
      v4f qb = *(const v4f*)(qrow + c0 + 4);
      v4f ra = *(const v4f*)(rrow + c0);
      v4f rb = *(const v4f*)(rrow + c0 + 4);
      v8h out;
#pragma unroll
      for (int j = 0; j < 8; ++j) {
        const int   c    = c0 + j;
        const float ang  = t * __expf((float)(c & 127) * NEG_LN);
        const float tri  = (c < 128) ? __sinf(ang) : __cosf(ang);
        const float base = (j < 4) ? (qa[j] + ra[j]) : (qb[j - 4] + rb[j - 4]);
        const float v    = base + tri;
        out[j]              = (_Float16)v;
        afrag[s][h2 * 8 + j] = (_Float16)v;
      }
      *(v8h*)(h16 + (long)tok * Dm + c0) = out;
    }
  }

  // struct head: logits partials, lane covers columns n == r (mod 16)
  float lpart[8];
#pragma unroll
  for (int g = 0; g < 8; ++g) lpart[g] = 0.f;

#pragma unroll 1
  for (int j = 0; j < 8; ++j) {  // n-tiles of 16 hidden units
    const int hrow = j * 16 + r;
    const _Float16* wrow = ws1a16 + (long)hrow * Dm;
    v8f c = {0.f, 0.f, 0.f, 0.f, 0.f, 0.f, 0.f, 0.f};
#pragma unroll
    for (int s = 0; s < 8; ++s) {
      v8h blo = *(const v8h*)(wrow + s * 32 + kh * 8);
      v8h bhi = *(const v8h*)(wrow + s * 32 + 16 + kh * 8);
      v16h b;
#pragma unroll
      for (int e = 0; e < 8; ++e) { b[e] = blo[e]; b[8 + e] = bhi[e]; }
      c = wmma_f16(afrag[s], b, c);
    }
    const float w2 = Ws2[hrow];
    const float hb = hvb[hrow];
#pragma unroll
    for (int g = 0; g < 8; ++g) {  // C layout: lane g-reg = row g + 8*kh
      float hv = c[g] + hb;
      hv = (hv > 0.f) ? hv : 0.2f * hv;
      lpart[g] += hv * w2;
    }
  }

  // butterfly reduce over the 16 lanes of each half (xor masks < 16 stay in half)
#pragma unroll
  for (int off = 1; off < 16; off <<= 1) {
#pragma unroll
    for (int g = 0; g < 8; ++g) lpart[g] += __shfl_xor(lpart[g], off, 32);
  }

  if (r == 0) {  // lanes 0 and 16 each write 8 rows of A_hat
    const float b2 = bs2[0];
#pragma unroll
    for (int g = 0; g < 8; ++g) {
      const int   m     = kh * 8 + g;
      const float logit = lpart[g] + b2;
      const float ah    = (1.f / (1.f + __expf(-100.f * logit))) * mask[tok0 + m];
      ahat[tok0 + m] = ah;
    }
  }
}

// ---------------------------------------------------------------------------
// K2: aggregator. Block = (b, 128-col u-tile) = 8 waves x 16 columns.
// The 16x256 f16 A-tile is staged once per block into LDS (double-buffered,
// via GLOBAL_LOAD_ASYNC_TO_LDS_B128 when the toolchain exposes it), then all
// 8 waves consume it with ds_load_b128. Wa B-fragments stay VGPR-resident.
// ---------------------------------------------------------------------------
__global__ __launch_bounds__(256) void aggregate_kernel(
    const _Float16* __restrict__ h16, const _Float16* __restrict__ wa16,
    const float* __restrict__ ahat, const float* __restrict__ ba,
    float* __restrict__ u_out) {
  __shared__ __align__(16) char ldsA[2 * 16 * LDSROW];

  const int tid  = threadIdx.x;
  const int wave = tid >> 5;
  const int lane = tid & 31;
  const int r    = lane & 15;
  const int kh   = lane >> 4;
  const int b    = blockIdx.y;
  const int ucol = blockIdx.x * 128 + wave * 16 + r;

  // preload B fragments (B 32x16 mirrors the A layout with N across lanes)
  const _Float16* wrow = wa16 + (long)ucol * Dm;
  v16h bfrag[8];
#pragma unroll
  for (int s = 0; s < 8; ++s) {
    v8h blo = *(const v8h*)(wrow + s * 32 + kh * 8);
    v8h bhi = *(const v8h*)(wrow + s * 32 + 16 + kh * 8);
#pragma unroll
    for (int e = 0; e < 8; ++e) { bfrag[s][e] = blo[e]; bfrag[s][8 + e] = bhi[e]; }
  }

  const float bau = ba[ucol];
  const float* Ab = ahat + (long)b * Ltok;
  const _Float16* Abase = h16 + (long)b * Ltok * Dm;

  // stage one 16x256 f16 tile (8 KB) into LDS buffer `buf`:
  // 512 chunks of 16B, 256 threads -> 2 chunks per thread.
  auto stage = [&](int mt, int buf) {
    const char* gsrc = (const char*)(Abase + (long)mt * 16 * Dm);
    char* dbase = ldsA + buf * (16 * LDSROW);
#pragma unroll
    for (int jj = 0; jj < 2; ++jj) {
      const int i   = tid + jj * 256;
      const int row = i >> 5;
      const int ch  = i & 31;
      const char* src = gsrc + row * 512 + ch * 16;
      char*       dst = dbase + row * LDSROW + ch * 16;
#if HAS_ASYNC
      __builtin_amdgcn_global_load_async_to_lds_b128(
          (as1_v4i*)(unsigned long long)src,
          (as3_v4i*)(unsigned)(unsigned long long)dst, 0, 0);
#else
      *(v8h*)dst = *(const v8h*)src;  // sync fallback: global load + ds_store
#endif
    }
  };

  stage(0, 0);
#if HAS_ASYNC
  __builtin_amdgcn_s_wait_asynccnt(0);
#endif
  __syncthreads();

  float acc = 0.f;
#pragma unroll 1
  for (int mt = 0; mt < Ltok / 16; ++mt) {
    const int buf = mt & 1;
    if (mt + 1 < Ltok / 16) stage(mt + 1, buf ^ 1);  // overlap copy with WMMAs

    const char* abase = ldsA + buf * (16 * LDSROW) + r * LDSROW;
    v8f c = {0.f, 0.f, 0.f, 0.f, 0.f, 0.f, 0.f, 0.f};
#pragma unroll
    for (int s = 0; s < 8; ++s) {
      v8h alo = *(const v8h*)(abase + s * 64 + kh * 16);
      v8h ahi = *(const v8h*)(abase + s * 64 + 32 + kh * 16);
      v16h a;
#pragma unroll
      for (int e = 0; e < 8; ++e) { a[e] = alo[e]; a[8 + e] = ahi[e]; }
      c = wmma_f16(a, bfrag[s], c);
    }

    v4f a0 = *(const v4f*)(Ab + mt * 16 + kh * 8);
    v4f a1 = *(const v4f*)(Ab + mt * 16 + kh * 8 + 4);
#pragma unroll
    for (int g = 0; g < 8; ++g) {  // row m = g + 8*kh of this tile
      float hv = c[g] + bau;
      hv = (hv > 0.f) ? hv : 0.2f * hv;
      const float aw = (g < 4) ? a0[g] : a1[g - 4];
      acc += hv * aw;  // sum over M (rows) folds into the accumulator layout
    }

    // copy of tile mt+1 must be LDS-visible to all waves before next iter;
    // the barrier also makes the ping-pong overwrite race-free.
#if HAS_ASYNC
    __builtin_amdgcn_s_wait_asynccnt(0);
#endif
    __syncthreads();
  }

  acc += __shfl_xor(acc, 16, 32);  // combine M 0..7 with M 8..15 halves
  if (kh == 0) u_out[(long)b * DUu + ucol] = acc;
}

// ---------------------------------------------------------------------------
extern "C" void kernel_launch(void* const* d_in, const int* in_sizes, int n_in,
                              void* d_out, int out_size, void* d_ws, size_t ws_size,
                              hipStream_t stream) {
  const int*   q_seq = (const int*)  d_in[0];
  const int*   r_seq = (const int*)  d_in[1];
  const float* t_seq = (const float*)d_in[2];
  const float* mask  = (const float*)d_in[3];
  const float* q_tab = (const float*)d_in[4];
  const float* r_tab = (const float*)d_in[5];
  const float* vnode = (const float*)d_in[6];
  const float* Ws1   = (const float*)d_in[7];
  const float* bs1   = (const float*)d_in[8];
  const float* Ws2   = (const float*)d_in[9];
  const float* bs2   = (const float*)d_in[10];
  const float* Wa    = (const float*)d_in[11];
  const float* ba    = (const float*)d_in[12];

  float* out   = (float*)d_out;
  float* u_out = out;                    // (64, 1024)
  float* ahat  = out + (size_t)Bb * DUu; // (64, 2048)

  // workspace layout (all 16B aligned): h16 | wa16 | ws1a16 | hvb
  char* w = (char*)d_ws;
  _Float16* h16    = (_Float16*)(w);
  size_t off = (size_t)NTOK * Dm * sizeof(_Float16);           // 67,108,864
  _Float16* wa16   = (_Float16*)(w + off);
  off += (size_t)DUu * Dm * sizeof(_Float16);                  // +524,288
  _Float16* ws1a16 = (_Float16*)(w + off);
  off += (size_t)Hh * Dm * sizeof(_Float16);                   // +65,536
  float*    hvb    = (float*)(w + off);                        // +512

  const int total = DUu * Dm + Hh * Dm;
  convert_weights<<<(total + 255) / 256, 256, 0, stream>>>(Wa, Ws1, wa16, ws1a16);
  virt_bias<<<1, Hh, 0, stream>>>(Ws1, bs1, vnode, hvb);

  prep_struct_kernel<<<NTOK / 16 / 8, 256, 0, stream>>>(
      q_seq, r_seq, t_seq, mask, q_tab, r_tab, ws1a16, hvb, Ws2, bs2, h16, ahat);

  aggregate_kernel<<<dim3(DUu / 128, Bb), 256, 0, stream>>>(h16, wa16, ahat, ba, u_out);
}
